// CrossAttention_6287832121664
// MI455X (gfx1250) — matrix operand order repaired and checked by reading, not compiled, not run on hardware
//
#include <hip/hip_runtime.h>
#include <hip/hip_bf16.h>
#include <stdint.h>

// ---------------------------------------------------------------------------
// Problem constants (reference: B,L,QD,H,D = 2,2048,1024,16,64)
// ---------------------------------------------------------------------------
static constexpr int CB  = 2;
static constexpr int CL  = 2048;
static constexpr int CQD = 1024;
static constexpr int CH  = 16;
static constexpr int CD  = 64;
static constexpr int CIN = CH * CD;        // 1024
static constexpr int CM  = CB * CL;        // 4096 rows

typedef __attribute__((ext_vector_type(16))) __bf16    v16bf;
typedef __attribute__((ext_vector_type(8)))  __bf16    v8bf;
typedef __attribute__((ext_vector_type(8)))  float     v8f;
typedef __attribute__((ext_vector_type(4)))  uint32_t  u32x4;
typedef __attribute__((ext_vector_type(8)))  uint32_t  u32x8;

#define WMMA_BF16(a, b, c) \
  __builtin_amdgcn_wmma_f32_16x16x32_bf16(false, (a), false, (b), (short)0, (c), false, false)

// ---------------------------------------------------------------------------
// Tensor Data Mover: 2D tile load Global -> LDS (D# per CDNA5 ISA ch.8).
//  - data_size = 2 bytes (bf16)
//  - tile_d0 elems per row (contiguous), tile_d1 rows, row stride in elems
//  - optional LDS padding (pad_ia/pad_am codes) to produce padded LDS strides
// Descriptor is wave-uniform; "s" constraints keep it in SGPRs.
// ---------------------------------------------------------------------------
static __device__ inline uint32_t lds_of(const void* p) {
  return (uint32_t)(unsigned long long)p;   // LDS aperture: addr[31:0] = LDS offset
}

static __device__ inline void tdm_load_2d(uint32_t lds_addr, const __bf16* gptr,
                                          uint32_t tile_d0, uint32_t tile_d1,
                                          uint64_t row_stride_elems,
                                          uint32_t pad_en, uint32_t pad_ia,
                                          uint32_t pad_am) {
  const uint64_t ga = (uint64_t)(unsigned long long)gptr;
  u32x4 g0;
  g0[0] = 1u;                                   // count=1, user descriptor
  g0[1] = lds_addr;                             // lds_addr [63:32]
  g0[2] = (uint32_t)ga;                         // global_addr [95:64]
  g0[3] = (uint32_t)((ga >> 32) & 0x01FFFFFFu)  // global_addr [120:96]
        | (2u << 30);                           // type=2 ("image") [127:126]
  u32x8 g1;
  g1[0] = (1u << 16)                            // data_size = 1 -> 2 bytes
        | (pad_en << 20) | (pad_ia << 22) | (pad_am << 25);
  g1[1] = (tile_d0 & 0xFFFFu) << 16;            // tensor_dim0[15:0]  (== tile_d0)
  g1[2] = ((tile_d0 >> 16) & 0xFFFFu)           // tensor_dim0[31:16]
        | ((tile_d1 & 0xFFFFu) << 16);          // tensor_dim1[15:0]  (== tile_d1)
  g1[3] = ((tile_d1 >> 16) & 0xFFFFu)           // tensor_dim1[31:16]
        | ((tile_d0 & 0xFFFFu) << 16);          // tile_dim0 [127:112]
  g1[4] = (tile_d1 & 0xFFFFu);                  // tile_dim1 [143:128]; tile_dim2=0
  g1[5] = (uint32_t)(row_stride_elems & 0xFFFFFFFFu);       // dim0_stride [191:160]
  g1[6] = (uint32_t)((row_stride_elems >> 32) & 0xFFFFu);   // dim0_stride [207:192]
  g1[7] = 0u;
  asm volatile("tensor_load_to_lds %0, %1" :: "s"(g0), "s"(g1) : "memory");
}

// ---------------------------------------------------------------------------
// Fragment loaders (wave32; contiguous k order, lane half h holds k = 16h..16h+15)
// ---------------------------------------------------------------------------
static __device__ inline v16bf ld_frag_contig16(const __bf16* p) {
  v8bf lo = *(const v8bf*)p;
  v8bf hi = *(const v8bf*)(p + 8);
  return __builtin_shufflevector(lo, hi, 0,1,2,3,4,5,6,7,8,9,10,11,12,13,14,15);
}

static __device__ inline v16bf ld_frag_A(const __bf16* row, int lane) {
  int off = (lane & 16) ? 16 : 0;
  v8bf lo = *(const v8bf*)(row + off);
  v8bf hi = *(const v8bf*)(row + off + 8);
  return __builtin_shufflevector(lo, hi, 0,1,2,3,4,5,6,7,8,9,10,11,12,13,14,15);
}

static __device__ inline float rmax16(float v) {
  v = fmaxf(v, __shfl_xor(v, 1));
  v = fmaxf(v, __shfl_xor(v, 2));
  v = fmaxf(v, __shfl_xor(v, 4));
  v = fmaxf(v, __shfl_xor(v, 8));
  return v;
}
static __device__ inline float rsum16(float v) {
  v += __shfl_xor(v, 1);
  v += __shfl_xor(v, 2);
  v += __shfl_xor(v, 4);
  v += __shfl_xor(v, 8);
  return v;
}

// ---------------------------------------------------------------------------
// fp32 -> bf16 conversion (grid-stride)
// ---------------------------------------------------------------------------
__global__ void cvt_kernel(const float* __restrict__ s, __bf16* __restrict__ d, int n) {
  for (int i = blockIdx.x * blockDim.x + threadIdx.x; i < n; i += gridDim.x * blockDim.x)
    d[i] = (__bf16)s[i];
}

// ---------------------------------------------------------------------------
// Tiled bf16 GEMM:  C[M,N] = A[M,K] * Bw[N,K]^T   (Bw row-major [N,K])
// Block tile 128x128, KC=32, 8 waves (4 M x 2 N), each wave 32x64 via 8 WMMAs.
// Tiles staged by the Tensor Data Mover, double-buffered: wave 0 issues the
// next iteration's DMA pair, waits tensorcnt<=2 (in-order => current pair
// done), all waves barrier, compute overlaps the in-flight DMA.
// LDS padded stride (40 elems = 32 data + 8 pad) produced by D# pad fields:
// pad_interval code 3 (16 DWORDs = 64B row), pad_amount code 3 (4 DWORDs = 16B).
// ---------------------------------------------------------------------------
#define GBM 128
#define GBN 128
#define GKC 32
#define GLDS 40

__global__ __launch_bounds__(256) void gemm_bf16(
    const __bf16* __restrict__ A, const __bf16* __restrict__ Bw,
    __bf16* __restrict__ Obf, float* __restrict__ Of,
    const float* __restrict__ bias, int M, int N, int K) {
  __shared__ __bf16 As[2][GBM * GLDS];
  __shared__ __bf16 Bs[2][GBN * GLDS];

  const int tid  = threadIdx.x;
  const int lane = tid & 31;
  const int wave = tid >> 5;
  const int wm   = wave & 3;
  const int wn   = wave >> 2;
  const int m0   = blockIdx.y * GBM;
  const int n0   = blockIdx.x * GBN;
  const int l15  = lane & 15;

  v8f acc[2][4];
  const v8f z8 = {0.f, 0.f, 0.f, 0.f, 0.f, 0.f, 0.f, 0.f};
#pragma unroll
  for (int i = 0; i < 2; ++i)
#pragma unroll
    for (int j = 0; j < 4; ++j) acc[i][j] = z8;

  const int nIter = K / GKC;
  if (wave == 0) {
    tdm_load_2d(lds_of(&As[0][0]), &A[(size_t)m0 * K],  GKC, GBM, (uint64_t)K, 1, 3, 3);
    tdm_load_2d(lds_of(&Bs[0][0]), &Bw[(size_t)n0 * K], GKC, GBN, (uint64_t)K, 1, 3, 3);
  }

  for (int it = 0; it < nIter; ++it) {
    const int buf = it & 1;
    if (wave == 0) {
      if (it + 1 < nIter) {
        const int kc2 = (it + 1) * GKC;
        tdm_load_2d(lds_of(&As[buf ^ 1][0]), &A[(size_t)m0 * K + kc2],  GKC, GBM, (uint64_t)K, 1, 3, 3);
        tdm_load_2d(lds_of(&Bs[buf ^ 1][0]), &Bw[(size_t)n0 * K + kc2], GKC, GBN, (uint64_t)K, 1, 3, 3);
        __builtin_amdgcn_s_wait_tensorcnt(2);   // current pair complete (in-order)
      } else {
        __builtin_amdgcn_s_wait_tensorcnt(0);
      }
    }
    __syncthreads();

    v16bf af[2];
#pragma unroll
    for (int i = 0; i < 2; ++i)
      af[i] = ld_frag_A(&As[buf][(wm * 32 + i * 16 + l15) * GLDS], lane);

    const int boff = (lane & 16) ? 16 : 0;
#pragma unroll
    for (int j = 0; j < 4; ++j) {
      v16bf bf_ = ld_frag_contig16(&Bs[buf][(wn * 64 + j * 16 + l15) * GLDS + boff]);
#pragma unroll
      for (int i = 0; i < 2; ++i) acc[i][j] = WMMA_BF16(af[i], bf_, acc[i][j]);
    }
    __syncthreads();
  }

  const int rbase = (lane & 16) ? 8 : 0;
#pragma unroll
  for (int i = 0; i < 2; ++i) {
#pragma unroll
    for (int j = 0; j < 4; ++j) {
      const int col = n0 + wn * 64 + j * 16 + l15;
#pragma unroll
      for (int r = 0; r < 8; ++r) {
        const int row = m0 + wm * 32 + i * 16 + rbase + r;
        const float v = acc[i][j][r];
        if (Of) Of[(size_t)row * N + col] = v + (bias ? bias[col] : 0.f);
        else    Obf[(size_t)row * N + col] = (__bf16)v;
      }
    }
  }
}

// ---------------------------------------------------------------------------
// Per-(b,l,h) RMSNorm (+scale) and FLUX RoPE on q,k; relayout q,k,v to
// [B,H,L,D] bf16 for attention.
// ---------------------------------------------------------------------------
__global__ __launch_bounds__(256) void norm_rope_kernel(
    const __bf16* __restrict__ q_raw,   // [M, INNER]
    const __bf16* __restrict__ kv_raw,  // [M, 2*INNER]
    const float* __restrict__ pe,       // [L, 32, 2, 2]
    const float* __restrict__ qsc, const float* __restrict__ ksc,
    __bf16* __restrict__ qh, __bf16* __restrict__ kh, __bf16* __restrict__ vh) {
  const int t = blockIdx.x * blockDim.x + threadIdx.x;
  if (t >= CB * CL * CH) return;
  const int h = t & (CH - 1);
  const int l = (t >> 4) & (CL - 1);
  const int b = t >> 4 >> 11;

  const size_t rowQ  = (size_t)(b * CL + l) * CIN + h * CD;
  const size_t rowKV = (size_t)(b * CL + l) * (2 * CIN);
  const float* pl = pe + (size_t)l * (CD / 2) * 4;
  const size_t outRow = ((size_t)(b * CH + h) * CL + l) * CD;

  float x[CD];

  // ---- q ----
  float ss = 0.f;
#pragma unroll
  for (int d = 0; d < CD; ++d) { x[d] = (float)q_raw[rowQ + d]; ss += x[d] * x[d]; }
  float rr = rsqrtf(ss * (1.0f / CD) + 1e-6f);
#pragma unroll
  for (int d = 0; d < CD; ++d) x[d] = x[d] * rr * qsc[d];
#pragma unroll
  for (int p = 0; p < CD / 2; ++p) {
    const float x0 = x[2 * p], x1 = x[2 * p + 1];
    qh[outRow + 2 * p]     = (__bf16)(pl[p * 4 + 0] * x0 + pl[p * 4 + 1] * x1);
    qh[outRow + 2 * p + 1] = (__bf16)(pl[p * 4 + 2] * x0 + pl[p * 4 + 3] * x1);
  }

  // ---- k ----
  ss = 0.f;
#pragma unroll
  for (int d = 0; d < CD; ++d) { x[d] = (float)kv_raw[rowKV + h * CD + d]; ss += x[d] * x[d]; }
  rr = rsqrtf(ss * (1.0f / CD) + 1e-6f);
#pragma unroll
  for (int d = 0; d < CD; ++d) x[d] = x[d] * rr * ksc[d];
#pragma unroll
  for (int p = 0; p < CD / 2; ++p) {
    const float x0 = x[2 * p], x1 = x[2 * p + 1];
    kh[outRow + 2 * p]     = (__bf16)(pl[p * 4 + 0] * x0 + pl[p * 4 + 1] * x1);
    kh[outRow + 2 * p + 1] = (__bf16)(pl[p * 4 + 2] * x0 + pl[p * 4 + 3] * x1);
  }

  // ---- v ----
#pragma unroll
  for (int d = 0; d < CD; ++d)
    vh[outRow + d] = kv_raw[rowKV + CIN + h * CD + d];
}

// ---------------------------------------------------------------------------
// Flash attention: one block per (b,h, 128 q rows). 8 waves x 16 q rows.
// K tile staged by TDM (wave 0), V^T staged by all threads concurrently.
// ---------------------------------------------------------------------------
#define AQT 128
#define VLDS 40

__global__ __launch_bounds__(256) void attn_kernel(
    const __bf16* __restrict__ qh, const __bf16* __restrict__ kh,
    const __bf16* __restrict__ vh, __bf16* __restrict__ oh) {
  __shared__ __bf16 Kt[32 * 64];        // [key][d]  (TDM destination)
  __shared__ __bf16 Vt[64 * VLDS];      // [d][key]  (transposed)
  __shared__ __bf16 Pw[8][16 * 32];     // per-wave P staging

  const int bh   = blockIdx.y;
  const int b    = bh >> 4;
  const int h    = bh & 15;
  const int q0   = blockIdx.x * AQT;
  const int tid  = threadIdx.x;
  const int lane = tid & 31;
  const int wave = tid >> 5;
  const int l15  = lane & 15;
  const int aoff = (lane & 16) ? 16 : 0;
  const int boff = (lane & 16) ? 16 : 0;

  const size_t headBase = (size_t)bh * CL * CD;

  v16bf qf[2];
  {
    const int qrow = q0 + wave * 16 + l15;
    const __bf16* qp = qh + headBase + (size_t)qrow * CD;
#pragma unroll
    for (int j = 0; j < 2; ++j) {
      v8bf lo = *(const v8bf*)(qp + j * 32 + aoff);
      v8bf hi = *(const v8bf*)(qp + j * 32 + aoff + 8);
      qf[j] = __builtin_shufflevector(lo, hi, 0,1,2,3,4,5,6,7,8,9,10,11,12,13,14,15);
    }
  }

  const v8f z8 = {0.f, 0.f, 0.f, 0.f, 0.f, 0.f, 0.f, 0.f};
  v8f oacc[4];
#pragma unroll
  for (int dt = 0; dt < 4; ++dt) oacc[dt] = z8;
  float mrow[8], lrow[8];
#pragma unroll
  for (int r = 0; r < 8; ++r) { mrow[r] = -1e30f; lrow[r] = 0.f; }

  const float sc = 0.125f;  // 1/sqrt(64)

  for (int kb = 0; kb < CL; kb += 32) {
    // ---- stage: K tile via Tensor Data Mover; V^T manually (all threads) ----
    if (wave == 0)
      tdm_load_2d(lds_of(&Kt[0]), kh + headBase + (size_t)kb * CD,
                  64, 32, 64ull, 0, 0, 0);
    {
      const int krow = tid >> 3;
      const int d8   = (tid & 7) * 8;
      const size_t g = headBase + (size_t)(kb + krow) * CD + d8;
      v8bf vv = *(const v8bf*)(vh + g);
#pragma unroll
      for (int e = 0; e < 8; ++e) Vt[(d8 + e) * VLDS + krow] = vv[e];
      if (kb + 32 < CL) __builtin_prefetch(vh + g + 32 * CD, 0, 1);
    }
    if (wave == 0) __builtin_amdgcn_s_wait_tensorcnt(0);
    __syncthreads();

    // ---- S = Q K^T (two 16-key sub-tiles), scaled ----
    v8f s[2];
#pragma unroll
    for (int st = 0; st < 2; ++st) {
      v8f a = z8;
#pragma unroll
      for (int j = 0; j < 2; ++j) {
        v16bf bfrag = ld_frag_contig16(&Kt[(st * 16 + l15) * 64 + j * 32 + boff]);
        a = WMMA_BF16(qf[j], bfrag, a);
      }
#pragma unroll
      for (int r = 0; r < 8; ++r) a[r] *= sc;
      s[st] = a;
    }

    // ---- online softmax over 32 keys ----
    float fac[8];
#pragma unroll
    for (int r = 0; r < 8; ++r) {
      float mx = rmax16(fmaxf(s[0][r], s[1][r]));
      const float mn = fmaxf(mrow[r], mx);
      fac[r] = __expf(mrow[r] - mn);
      mrow[r] = mn;
      const float p0 = __expf(s[0][r] - mn);
      const float p1 = __expf(s[1][r] - mn);
      s[0][r] = p0; s[1][r] = p1;
      lrow[r] = lrow[r] * fac[r] + rsum16(p0 + p1);
    }
#pragma unroll
    for (int dt = 0; dt < 4; ++dt)
#pragma unroll
      for (int r = 0; r < 8; ++r) oacc[dt][r] *= fac[r];

    // ---- P (C-layout) -> LDS, re-read in A-layout ----
    {
      const int rb = (lane & 16) ? 8 : 0;
#pragma unroll
      for (int st = 0; st < 2; ++st)
#pragma unroll
        for (int r = 0; r < 8; ++r)
          Pw[wave][(rb + r) * 32 + st * 16 + l15] = (__bf16)s[st][r];
    }
    __syncthreads();

    // ---- O += P V ----
    {
      v16bf pf = ld_frag_A(&Pw[wave][l15 * 32], lane);
#pragma unroll
      for (int dt = 0; dt < 4; ++dt) {
        v16bf vfrag = ld_frag_contig16(&Vt[(dt * 16 + l15) * VLDS + boff]);
        oacc[dt] = WMMA_BF16(pf, vfrag, oacc[dt]);
      }
    }
    __syncthreads();
  }

  const int rbase = (lane & 16) ? 8 : 0;
#pragma unroll
  for (int r = 0; r < 8; ++r) {
    const float inv = 1.0f / lrow[r];
    const int row = q0 + wave * 16 + rbase + r;
    const size_t orow = ((size_t)b * CL + row) * CIN + h * CD;
#pragma unroll
    for (int dt = 0; dt < 4; ++dt)
      oh[orow + dt * 16 + l15] = (__bf16)(oacc[dt][r] * inv);
  }
}

// ---------------------------------------------------------------------------
// Host launcher
// ---------------------------------------------------------------------------
extern "C" void kernel_launch(void* const* d_in, const int* in_sizes, int n_in,
                              void* d_out, int out_size, void* d_ws, size_t ws_size,
                              hipStream_t stream) {
  const float* x     = (const float*)d_in[0];
  const float* pe    = (const float*)d_in[1];
  const float* Wq    = (const float*)d_in[2];
  const float* Wkv   = (const float*)d_in[3];
  const float* Wproj = (const float*)d_in[4];
  const float* bproj = (const float*)d_in[5];
  const float* qsc   = (const float*)d_in[6];
  const float* ksc   = (const float*)d_in[7];
  float* out = (float*)d_out;

  char* ws = (char*)d_ws;
  size_t off = 0;
  auto alloc = [&](size_t bytes) { void* p = ws + off; off += (bytes + 255) & ~size_t(255); return p; };
  __bf16* x_bf     = (__bf16*)alloc((size_t)CM * CQD * 2);
  __bf16* Wq_bf    = (__bf16*)alloc((size_t)CIN * CQD * 2);
  __bf16* Wkv_bf   = (__bf16*)alloc((size_t)2 * CIN * CQD * 2);
  __bf16* Wproj_bf = (__bf16*)alloc((size_t)CQD * CIN * 2);
  __bf16* q_raw    = (__bf16*)alloc((size_t)CM * CIN * 2);
  __bf16* kv_raw   = (__bf16*)alloc((size_t)CM * 2 * CIN * 2);
  __bf16* qh       = (__bf16*)alloc((size_t)CB * CH * CL * CD * 2);
  __bf16* kh       = (__bf16*)alloc((size_t)CB * CH * CL * CD * 2);
  __bf16* vh       = (__bf16*)alloc((size_t)CB * CH * CL * CD * 2);
  __bf16* oh       = (__bf16*)alloc((size_t)CM * CIN * 2);

  cvt_kernel<<<2048, 256, 0, stream>>>(x,     x_bf,     CM * CQD);
  cvt_kernel<<<1024, 256, 0, stream>>>(Wq,    Wq_bf,    CIN * CQD);
  cvt_kernel<<<2048, 256, 0, stream>>>(Wkv,   Wkv_bf,   2 * CIN * CQD);
  cvt_kernel<<<1024, 256, 0, stream>>>(Wproj, Wproj_bf, CQD * CIN);

  gemm_bf16<<<dim3(CIN / GBN, CM / GBM), 256, 0, stream>>>(
      x_bf, Wq_bf, q_raw, nullptr, nullptr, CM, CIN, CQD);
  gemm_bf16<<<dim3(2 * CIN / GBN, CM / GBM), 256, 0, stream>>>(
      x_bf, Wkv_bf, kv_raw, nullptr, nullptr, CM, 2 * CIN, CQD);

  norm_rope_kernel<<<(CB * CL * CH + 255) / 256, 256, 0, stream>>>(
      q_raw, kv_raw, pe, qsc, ksc, qh, kh, vh);

  attn_kernel<<<dim3(CL / AQT, CB * CH), 256, 0, stream>>>(qh, kh, vh, oh);

  gemm_bf16<<<dim3(CQD / GBN, CM / GBM), 256, 0, stream>>>(
      oh, Wproj_bf, nullptr, out, bproj, CM, CQD, CIN);
}
